// Net_16793322127662
// MI455X (gfx1250) — matrix operand order repaired and checked by reading, not compiled, not run on hardware
//
#include <hip/hip_runtime.h>
#include <hip/hip_bf16.h>
#include <math.h>

typedef __attribute__((ext_vector_type(16))) _Float16 v16h;
typedef __attribute__((ext_vector_type(8)))  _Float16 v8h;
typedef __attribute__((ext_vector_type(8)))  float    v8f;

#define CL    32
#define NOBJ  128
#define NBAT  16
#define NROLL 8

// Pre-swizzled weight blob: 34 fragments x 32 lanes x 16 halves (f16),
// then 260 floats of biases/at2. rc0/at0 carry their bias in K-row 67
// (comb[67] == 1), so GEMM1 needs no C-splat.
#define NFRAG       34
#define FRAG_RC0    0   // 12 frags (nt*3 + kt), K padded 67->96, row67 = bias
#define FRAG_AT0    12  // 12 frags, row67 = bias
#define FRAG_RC1    24  // 4 frags (nt*2 + kt)
#define FRAG_AT1    28  // 4 frags
#define FRAG_RC2    32  // 2 frags (nt)
#define WFRAG_BYTES (NFRAG * 512 * 2)   // 34816
#define BIAS_FLOATS 260
#define BLOB_BYTES  (WFRAG_BYTES + BIAS_FLOATS * 4)  // 35856 (16B multiple)
#define BLOB_U4     (BLOB_BYTES / 16)                // 2241

// bias blob offsets (floats)
#define B_RC1 128
#define B_AT1 160
#define B_RC2 192
#define B_AT2W 224
#define B_AT2B 256

__device__ __forceinline__ void cbar() { asm volatile("" ::: "memory"); }

__device__ __forceinline__ v8f wmma_f16(v16h a, v16h b, v8f c) {
  return __builtin_amdgcn_wmma_f32_16x16x32_f16(false, a, false, b, (short)0, c,
                                                false, false);
}

__device__ __forceinline__ v16h cat16(v8h lo, v8h hi) {
  return __builtin_shufflevector(lo, hi, 0, 1, 2, 3, 4, 5, 6, 7,
                                 8, 9, 10, 11, 12, 13, 14, 15);
}

// B fragment from pre-swizzled blob: lane's 16 halves contiguous (2x b128).
__device__ __forceinline__ v16h b_frag(const _Float16* WF, int fid, int lane) {
  const _Float16* p = WF + (fid << 9) + (lane << 4);
  return cat16(*(const v8h*)p, *(const v8h*)(p + 8));
}

// A fragment from row-major f16 tile (row stride 64 halves).
__device__ __forceinline__ v16h a_frag(const _Float16* T, int ktile, int lane) {
  const int row = lane & 15;
  const int hi  = (lane & 16) ? 8 : 0;
  const _Float16* p = T + row * 64 + ktile * 32 + hi;
  return cat16(*(const v8h*)p, *(const v8h*)(p + 16));
}

// ---------------------------------------------------------------------------
// One-time weight pre-swizzle (bias folded into rc0/at0 K-row 67).
// ---------------------------------------------------------------------------
__global__ void prep_kernel(const float* __restrict__ rc0_w, const float* __restrict__ rc0_b,
                            const float* __restrict__ rc1_w, const float* __restrict__ rc1_b,
                            const float* __restrict__ rc2_w, const float* __restrict__ rc2_b,
                            const float* __restrict__ at0_w, const float* __restrict__ at0_b,
                            const float* __restrict__ at1_w, const float* __restrict__ at1_b,
                            const float* __restrict__ at2_w, const float* __restrict__ at2_b,
                            _Float16* __restrict__ gWF, float* __restrict__ gBB) {
  const int gid = blockIdx.x * blockDim.x + threadIdx.x;
  if (gid < NFRAG * 512) {
    const int fid   = gid >> 9;
    const int r     = gid & 511;
    const int lane  = r >> 4;
    const int e     = r & 15;
    const int khalf = ((lane & 16) ? 8 : 0) + ((e & 8) ? 8 : 0);
    const int ncol  = lane & 15;
    float v = 0.f;
    if (fid < 12) {                       // rc0: 67x64 (+bias row 67)
      const int nt = fid / 3, kt = fid % 3;
      const int k = kt * 32 + khalf + e, n = nt * 16 + ncol;
      v = (k < 67) ? rc0_w[k * 64 + n] : (k == 67 ? rc0_b[n] : 0.f);
    } else if (fid < 24) {                // at0 (+bias row 67)
      const int f = fid - FRAG_AT0, nt = f / 3, kt = f % 3;
      const int k = kt * 32 + khalf + e, n = nt * 16 + ncol;
      v = (k < 67) ? at0_w[k * 64 + n] : (k == 67 ? at0_b[n] : 0.f);
    } else if (fid < 28) {                // rc1: 64x32
      const int f = fid - FRAG_RC1, nt = f / 2, kt = f % 2;
      const int k = kt * 32 + khalf + e, n = nt * 16 + ncol;
      v = rc1_w[k * 32 + n];
    } else if (fid < 32) {                // at1: 64x32
      const int f = fid - FRAG_AT1, nt = f / 2, kt = f % 2;
      const int k = kt * 32 + khalf + e, n = nt * 16 + ncol;
      v = at1_w[k * 32 + n];
    } else {                              // rc2: 32x32
      const int nt = fid - FRAG_RC2;
      const int k = khalf + e, n = nt * 16 + ncol;
      v = rc2_w[k * 32 + n];
    }
    gWF[gid] = (_Float16)v;
  }
  if (gid < BIAS_FLOATS) {
    float v = 0.f;
    if (gid >= 128 && gid < 160)      v = rc1_b[gid - 128];
    else if (gid >= 160 && gid < 192) v = at1_b[gid - 160];
    else if (gid >= 192 && gid < 224) v = rc2_b[gid - 192];
    else if (gid >= 224 && gid < 256) v = at2_w[gid - 224];
    else if (gid == 256)              v = at2_b[0];
    gBB[gid] = v;
  }
}

// ---------------------------------------------------------------------------
// State encoder + present_states (== x) copy. Writes f32 and f16 state.
// ---------------------------------------------------------------------------
__global__ void init_kernel(const float* __restrict__ x,
                            const float* __restrict__ se_w,
                            const float* __restrict__ se_b,
                            float* __restrict__ S0, _Float16* __restrict__ S016,
                            float* __restrict__ out_present) {
  const int idx = blockIdx.x * blockDim.x + threadIdx.x;
  if (idx < NBAT * 4 * NOBJ * 4) out_present[idx] = x[idx];
  if (idx < NBAT * NOBJ) {
    const int b = idx >> 7, i = idx & 127;
    const float* xr = x + (((b * 4) + 3) * NOBJ + i) * 4;
    float* s = S0 + idx * CL;
    _Float16* s16 = S016 + idx * CL;
#pragma unroll
    for (int c = 0; c < 4; ++c) { s[c] = xr[c]; s16[c] = (_Float16)xr[c]; }
#pragma unroll
    for (int c = 4; c < CL; ++c) {
      float a = se_b[c];
#pragma unroll
      for (int k = 0; k < 4; ++k) a += xr[k] * se_w[k * CL + c];
      s[c] = a; s16[c] = (_Float16)a;
    }
  }
}

// ---------------------------------------------------------------------------
// One rollout step. 256 blocks; each of the 8 waves owns ONE object i and
// loops over all 8 j-tiles, accumulating rel_dyn wave-privately in registers
// (no atomics, no inter-wave sync). Then the same wave runs the per-object
// tail for its i. ~272 WMMAs per wave, LDS fill amortized over 8 objects.
// ---------------------------------------------------------------------------
__global__ __launch_bounds__(256)
void step_kernel(const float* __restrict__ S_in, const _Float16* __restrict__ S16_in,
                 float* __restrict__ S_out, _Float16* __restrict__ S16_out,
                 const uint4* __restrict__ gblob,
                 float* __restrict__ out_roll, int step,
                 const float* __restrict__ sc0_w, const float* __restrict__ sc0_b,
                 const float* __restrict__ sc1_w, const float* __restrict__ sc1_b,
                 const float* __restrict__ af0_w, const float* __restrict__ af0_b,
                 const float* __restrict__ af1_w, const float* __restrict__ af1_b,
                 const float* __restrict__ af2_w, const float* __restrict__ af2_b,
                 const float* __restrict__ o0_w,  const float* __restrict__ o0_b,
                 const float* __restrict__ o1_w,  const float* __restrict__ o1_b) {
  __shared__ uint4 wblob4[BLOB_U4];     // fragments + biases (35856 B)
  __shared__ uint4 sblob4[512];         // batch state f16 (8192 B)
  __shared__ uint4 stage4[8][128];      // per-wave 16x64 f16 staging (16384 B)
  __shared__ float attrow[8][16];
  __shared__ float relsum[8][CL];
  __shared__ float tmem[8][3][CL];

  const int tid  = threadIdx.x;
  const int lane = tid & 31;
  const int wave = tid >> 5;
  const int b    = blockIdx.x >> 4;           // 16 i-groups per batch
  const int i    = (blockIdx.x & 15) * 8 + wave;
  const float* Sb = S_in + b * NOBJ * CL;

  // ---- cooperative LDS init (128-bit copies) ----
  for (int idx = tid; idx < BLOB_U4; idx += 256) wblob4[idx] = gblob[idx];
  {
    const uint4* gs16 = (const uint4*)(S16_in + b * NOBJ * CL);
    for (int idx = tid; idx < 512; idx += 256) sblob4[idx] = gs16[idx];
  }
  __syncthreads();

  const _Float16* WF   = (const _Float16*)wblob4;
  const float*    BB   = (const float*)((const char*)wblob4 + WFRAG_BYTES);
  const _Float16* Sf16 = (const _Float16*)sblob4;
  _Float16*       st   = (_Float16*)stage4[wave];

  const int row  = lane & 15;
  const int hi   = (lane & 16) ? 8 : 0;
  const int ncol = lane & 15;

  // s_i half of comb (constant over the j loop)
  const _Float16* pi = Sf16 + i * CL + hi;
  const v16h acomb0 = cat16(*(const v8h*)pi, *(const v8h*)(pi + 16));
  const float si0 = Sb[i * CL + 0], si1 = Sb[i * CL + 1];

  float acc0 = 0.f, acc1 = 0.f;   // wave-private rel_dyn partials

#pragma unroll 1
  for (int mt = 0; mt < 8; ++mt) {
    const int j = mt * 16 + row;
    const _Float16* pj = Sf16 + j * CL + hi;
    const v16h acomb1 = cat16(*(const v8h*)pj, *(const v8h*)(pj + 16));
    const float sj0 = Sb[j * CL + 0], sj1 = Sb[j * CL + 1];
    const float dx = si0 - sj0, dy = si1 - sj1;
    const float dist = sqrtf(dx * dx + dy * dy + 1e-10f);
    v16h acomb2;
#pragma unroll
    for (int e = 0; e < 16; ++e) acomb2[e] = (_Float16)0.f;
    if (hi == 0) {                 // K=64..67 live in the low K-half
      acomb2[0] = (_Float16)dist;
      acomb2[1] = (_Float16)dx;
      acomb2[2] = (_Float16)dy;
      acomb2[3] = (_Float16)1.f;   // bias feature (weight row 67)
    }

    // ---- relation chain ----
    // r1 = relu(comb @ rc0+bias) : 16x64, C starts at 0
#pragma unroll
    for (int nt = 0; nt < 4; ++nt) {
      v8f c;
#pragma unroll
      for (int r = 0; r < 8; ++r) c[r] = 0.f;
      c = wmma_f16(acomb0, b_frag(WF, FRAG_RC0 + nt * 3 + 0, lane), c);
      c = wmma_f16(acomb1, b_frag(WF, FRAG_RC0 + nt * 3 + 1, lane), c);
      c = wmma_f16(acomb2, b_frag(WF, FRAG_RC0 + nt * 3 + 2, lane), c);
      const int n = nt * 16 + ncol;
#pragma unroll
      for (int r = 0; r < 8; ++r) st[(hi + r) * 64 + n] = (_Float16)fmaxf(c[r], 0.f);
    }
    cbar();
    // r2 = relu(r1 @ rc1 + b1) : 16x32 (keep f32 copy for residual)
    v8f r2f0, r2f1;
    {
      const v16h a0 = a_frag(st, 0, lane);
      const v16h a1 = a_frag(st, 1, lane);
#pragma unroll
      for (int nt = 0; nt < 2; ++nt) {
        v8f c;
        const float bv = BB[B_RC1 + nt * 16 + ncol];
#pragma unroll
        for (int r = 0; r < 8; ++r) c[r] = bv;
        c = wmma_f16(a0, b_frag(WF, FRAG_RC1 + nt * 2 + 0, lane), c);
        c = wmma_f16(a1, b_frag(WF, FRAG_RC1 + nt * 2 + 1, lane), c);
        const int n = nt * 16 + ncol;
#pragma unroll
        for (int r = 0; r < 8; ++r) {
          const float rv = fmaxf(c[r], 0.f);
          c[r] = rv;
          st[(hi + r) * 64 + n] = (_Float16)rv;
        }
        if (nt == 0) r2f0 = c; else r2f1 = c;
      }
    }
    cbar();
    // rel = r2 @ rc2 + b2 + r2 (residual from f32 registers)
    v8f rel0, rel1;
    {
      const v16h a0 = a_frag(st, 0, lane);
#pragma unroll
      for (int nt = 0; nt < 2; ++nt) {
        v8f c;
        const float bv = BB[B_RC2 + nt * 16 + ncol];
#pragma unroll
        for (int r = 0; r < 8; ++r) c[r] = bv;
        c = wmma_f16(a0, b_frag(WF, FRAG_RC2 + nt, lane), c);
#pragma unroll
        for (int r = 0; r < 8; ++r) c[r] += (nt == 0) ? r2f0[r] : r2f1[r];
        if (nt == 0) rel0 = c; else rel1 = c;
      }
    }

    // ---- attention chain ----
#pragma unroll
    for (int nt = 0; nt < 4; ++nt) {
      v8f c;
#pragma unroll
      for (int r = 0; r < 8; ++r) c[r] = 0.f;
      c = wmma_f16(acomb0, b_frag(WF, FRAG_AT0 + nt * 3 + 0, lane), c);
      c = wmma_f16(acomb1, b_frag(WF, FRAG_AT0 + nt * 3 + 1, lane), c);
      c = wmma_f16(acomb2, b_frag(WF, FRAG_AT0 + nt * 3 + 2, lane), c);
      const int n = nt * 16 + ncol;
#pragma unroll
      for (int r = 0; r < 8; ++r) st[(hi + r) * 64 + n] = (_Float16)tanhf(c[r]);
    }
    cbar();
    {
      const v16h a0 = a_frag(st, 0, lane);
      const v16h a1 = a_frag(st, 1, lane);
#pragma unroll
      for (int nt = 0; nt < 2; ++nt) {
        v8f c;
        const float bv = BB[B_AT1 + nt * 16 + ncol];
#pragma unroll
        for (int r = 0; r < 8; ++r) c[r] = bv;
        c = wmma_f16(a0, b_frag(WF, FRAG_AT1 + nt * 2 + 0, lane), c);
        c = wmma_f16(a1, b_frag(WF, FRAG_AT1 + nt * 2 + 1, lane), c);
        const int n = nt * 16 + ncol;
#pragma unroll
        for (int r = 0; r < 8; ++r) st[(hi + r) * 64 + n] = (_Float16)tanhf(c[r]);
      }
    }
    cbar();
    // att3 = exp(att2 @ at2 + b) per row
    if (lane < 16) {
      const _Float16* pr = st + lane * 64;
      float s = BB[B_AT2B];
#pragma unroll
      for (int q = 0; q < 4; ++q) {
        const v8h v = *(const v8h*)(pr + q * 8);
#pragma unroll
        for (int e = 0; e < 8; ++e) s += (float)v[e] * BB[B_AT2W + q * 8 + e];
      }
      attrow[wave][lane] = expf(s);
    }
    cbar();

    // diag-masked attention-weighted accumulate (register-resident)
#pragma unroll
    for (int r = 0; r < 8; ++r) {
      const int jj = mt * 16 + hi + r;
      const float f = (jj == i) ? 0.f : attrow[wave][hi + r];
      acc0 += rel0[r] * f;
      acc1 += rel1[r] * f;
    }
    cbar();
  }

  // combine the two M-halves; lanes 0..15 hold channel sums
  acc0 += __shfl_xor(acc0, 16, 32);
  acc1 += __shfl_xor(acc1, 16, 32);
  if (lane < 16) {
    relsum[wave][lane]      = acc0;
    relsum[wave][16 + lane] = acc1;
  }
  cbar();

  // ---- per-object tail (same wave; lane = output channel c) ----
  {
    const int c = lane;
    float* tA = tmem[wave][0];
    float* tB = tmem[wave][1];
    float* tC = tmem[wave][2];
    float si[CL];
#pragma unroll
    for (int k = 0; k < CL; ++k) si[k] = Sb[i * CL + k];
    float a = sc0_b[c];
#pragma unroll
    for (int k = 0; k < CL; ++k) a += si[k] * sc0_w[k * CL + c];
    const float h1 = fmaxf(a, 0.f);
    tA[c] = h1;
    cbar();
    a = sc1_b[c];
#pragma unroll
    for (int k = 0; k < CL; ++k) a += tA[k] * sc1_w[k * CL + c];
    const float dyn = a + h1 + relsum[wave][c];
    tB[c] = dyn;
    cbar();
    a = af0_b[c];
#pragma unroll
    for (int k = 0; k < CL; ++k) a += tB[k] * af0_w[k * CL + c];
    const float aff1 = tanhf(a);
    tA[c] = aff1;
    cbar();
    a = af1_b[c];
#pragma unroll
    for (int k = 0; k < CL; ++k) a += tA[k] * af1_w[k * CL + c];
    const float aff2 = tanhf(a) + aff1;
    tB[c] = aff2;
    cbar();
    a = af2_b[c];
#pragma unroll
    for (int k = 0; k < CL; ++k) a += tB[k] * af2_w[k * CL + c];
    tC[c] = a;
    cbar();
    a = o0_b[c];
#pragma unroll
    for (int k = 0; k < CL; ++k) a += tC[k] * o0_w[k * CL + c];
#pragma unroll
    for (int k = 0; k < CL; ++k) a += si[k] * o0_w[(CL + k) * CL + c];
    const float out1 = tanhf(a);
    tA[c] = out1;
    cbar();
    a = o1_b[c];
#pragma unroll
    for (int k = 0; k < CL; ++k) a += tA[k] * o1_w[k * CL + c];
    float res = a + out1;
    if (c < 2) res += si[c];
    S_out[(b * NOBJ + i) * CL + c] = res;
    S16_out[(b * NOBJ + i) * CL + c] = (_Float16)res;
    if (c < 4) out_roll[(((b * NROLL) + step) * NOBJ + i) * 4 + c] = res;
  }
}

// ---------------------------------------------------------------------------
extern "C" void kernel_launch(void* const* d_in, const int* in_sizes, int n_in,
                              void* d_out, int out_size, void* d_ws, size_t ws_size,
                              hipStream_t stream) {
  (void)in_sizes; (void)n_in; (void)out_size; (void)ws_size;
  const float* x     = (const float*)d_in[0];
  const float* se_w  = (const float*)d_in[1];
  const float* se_b  = (const float*)d_in[2];
  const float* sc0_w = (const float*)d_in[3];
  const float* sc0_b = (const float*)d_in[4];
  const float* sc1_w = (const float*)d_in[5];
  const float* sc1_b = (const float*)d_in[6];
  const float* rc0_w = (const float*)d_in[7];
  const float* rc0_b = (const float*)d_in[8];
  const float* rc1_w = (const float*)d_in[9];
  const float* rc1_b = (const float*)d_in[10];
  const float* rc2_w = (const float*)d_in[11];
  const float* rc2_b = (const float*)d_in[12];
  const float* at0_w = (const float*)d_in[13];
  const float* at0_b = (const float*)d_in[14];
  const float* at1_w = (const float*)d_in[15];
  const float* at1_b = (const float*)d_in[16];
  const float* at2_w = (const float*)d_in[17];
  const float* at2_b = (const float*)d_in[18];
  const float* af0_w = (const float*)d_in[19];
  const float* af0_b = (const float*)d_in[20];
  const float* af1_w = (const float*)d_in[21];
  const float* af1_b = (const float*)d_in[22];
  const float* af2_w = (const float*)d_in[23];
  const float* af2_b = (const float*)d_in[24];
  const float* o0_w  = (const float*)d_in[25];
  const float* o0_b  = (const float*)d_in[26];
  const float* o1_w  = (const float*)d_in[27];
  const float* o1_b  = (const float*)d_in[28];
  // d_in[29] = num_rollout (fixed reference constant == NROLL)

  float* out      = (float*)d_out;
  float* out_roll = out;                             // (16,8,128,4)
  float* out_pres = out + NBAT * NROLL * NOBJ * 4;   // (16,4,128,4) == x

  const size_t SFLOATS = (size_t)NBAT * NOBJ * CL;   // 65536
  float*     stateA = (float*)d_ws;
  float*     stateB = stateA + SFLOATS;
  _Float16*  st16A  = (_Float16*)(stateB + SFLOATS);
  _Float16*  st16B  = st16A + SFLOATS;
  char*      blob   = (char*)(st16B + SFLOATS);
  _Float16*  gWF    = (_Float16*)blob;
  float*     gBB    = (float*)(blob + WFRAG_BYTES);

  prep_kernel<<<(NFRAG * 512 + 255) / 256, 256, 0, stream>>>(
      rc0_w, rc0_b, rc1_w, rc1_b, rc2_w, rc2_b,
      at0_w, at0_b, at1_w, at1_b, at2_w, at2_b, gWF, gBB);

  init_kernel<<<(NBAT * 4 * NOBJ * 4 + 255) / 256, 256, 0, stream>>>(
      x, se_w, se_b, stateA, st16A, out_pres);

  float* Sin = stateA;      float* Sout = stateB;
  _Float16* S16in = st16A;  _Float16* S16out = st16B;
  for (int step = 0; step < NROLL; ++step) {
    step_kernel<<<(NBAT * NOBJ) / 8, 256, 0, stream>>>(
        Sin, S16in, Sout, S16out, (const uint4*)blob, out_roll, step,
        sc0_w, sc0_b, sc1_w, sc1_b,
        af0_w, af0_b, af1_w, af1_b, af2_w, af2_b,
        o0_w, o0_b, o1_w, o1_b);
    float* t = Sin; Sin = Sout; Sout = t;
    _Float16* t16 = S16in; S16in = S16out; S16out = t16;
  }
}
